// CausalSelfAttention_31525059953201
// MI455X (gfx1250) — matrix operand order repaired and checked by reading, not compiled, not run on hardware
//
#include <hip/hip_runtime.h>

// CDNA5 / gfx1250 causal self-attention forward.
// B=4, T=2048, C=1024, H=16, D=64.
// bf16 WMMA (v_wmma_f32_16x16x32_bf16), fp32 accumulation, base-2 online
// softmax (log2e folded into the Q scale -> bare v_exp_f32).
// Fragments are pairs of b128 loads (CDNA5 fragment maps are per-lane
// contiguous); 32 rows per wave so B-fragments amortize over two row groups.
// Attention K/V staging uses GLOBAL_LOAD_ASYNC_TO_LDS_B128 (ASYNCcnt) with
// double-buffered LDS tiles when the toolchain declares the builtin.
// Workspace: Q,K [B,H,T,D] bf16, V [B,H,D,T] bf16 (transposed), Y [B,T,C] bf16.

typedef __bf16 v16bf __attribute__((ext_vector_type(16)));
typedef __bf16 v2bf  __attribute__((ext_vector_type(2)));
typedef float  v8f   __attribute__((ext_vector_type(8)));
typedef float  v2f   __attribute__((ext_vector_type(2)));
typedef int    v4i   __attribute__((ext_vector_type(4)));

union frag16 { v16bf v; uint4 u[2]; };

// A-matrix fragment from row-major [m][k] storage; `row` -> &tile[m][kk].
__device__ __forceinline__ v16bf a_frag(const __bf16* row, int half) {
  frag16 f;
  f.u[0] = *(const uint4*)(row + 8 * half);        // elems 0..7  (k = 8h..)
  f.u[1] = *(const uint4*)(row + 16 + 8 * half);   // elems 8..15 (k = 16+8h..)
  return f.v;
}
// B-matrix fragment from n-major [n][k] storage; `row` -> &tileT[n][kk].
__device__ __forceinline__ v16bf b_frag(const __bf16* row, int half) {
  frag16 f;
  f.u[0] = *(const uint4*)(row + 8 * half);
  f.u[1] = *(const uint4*)(row + 16 + 8 * half);
  return f.v;
}

__device__ __forceinline__ v8f wmma_bf16(v16bf a, v16bf b, v8f c) {
  return __builtin_amdgcn_wmma_f32_16x16x32_bf16(
      false, a, false, b, (short)0, c, false, false);
}

// Packed f32->bf16 conversion (lowers to v_cvt_pk_bf16_f32).
__device__ __forceinline__ unsigned pack2(float a, float b) {
  v2f f; f[0] = a; f[1] = b;
  union { v2bf h; unsigned u; } pk;
  pk.h = __builtin_convertvector(f, v2bf);
  return pk.u;
}
__device__ __forceinline__ uint2 pack4(float a, float b, float c, float d) {
  uint2 r; r.x = pack2(a, b); r.y = pack2(c, d);
  return r;
}

// ---- Async global->LDS copy (CDNA5 ASYNCcnt path) with safe fallback ------
// Signature (from toolchain): (v4i addrspace(1)*, v4i addrspace(3)*, imm, imm)
#if __has_builtin(__builtin_amdgcn_global_load_async_to_lds_b128)
#define ATTN_ASYNC_LDS 1
__device__ __forceinline__ void copy16(__bf16* dst, const __bf16* src) {
  __builtin_amdgcn_global_load_async_to_lds_b128(
      (__attribute__((address_space(1))) v4i*)src,
      (__attribute__((address_space(3))) v4i*)dst, 0, 0);
}
#else
#define ATTN_ASYNC_LDS 0
__device__ __forceinline__ void copy16(__bf16* dst, const __bf16* src) {
  *(uint4*)dst = *(const uint4*)src;
}
#endif
__device__ __forceinline__ void wait_async() {
#if ATTN_ASYNC_LDS
#if __has_builtin(__builtin_amdgcn_s_wait_asynccnt)
  __builtin_amdgcn_s_wait_asynccnt(0);
#else
  asm volatile("s_wait_asynccnt 0" ::: "memory");
#endif
#endif
}

// 1/sqrt(64) * log2(e): scores come out in log2 domain -> exp2 softmax.
#define QSCALE (0.125f * 1.4426950408889634f)

// ===========================================================================
// Kernel 1: qkv = x @ w_qkv + b_qkv ; scatter to Q,K [B,H,T,D], V [B,H,D,T].
// Q pre-scaled by QSCALE. Grid (M/128, 3C/64), block 128 (4 waves x 32 rows).
// ===========================================================================
__global__ __launch_bounds__(128)
void qkv_gemm_kernel(const float* __restrict__ x, const float* __restrict__ w,
                     const float* __restrict__ bias,
                     __bf16* __restrict__ Qo, __bf16* __restrict__ Ko,
                     __bf16* __restrict__ Vo) {
  constexpr int Kd = 1024, N3 = 3072, T = 2048, H = 16, D = 64;
  __shared__ __align__(16) __bf16 Asm[128 * 40];   // [m][k], stride 40 (80B)
  __shared__ __align__(16) __bf16 BsmT[64 * 40];   // [n][k], stride 40

  const int tid = threadIdx.x, wave = tid >> 5, lane = tid & 31;
  const int half = lane >> 4, ncol = lane & 15;
  const int mBase = blockIdx.x * 128, nBase = blockIdx.y * 64;

  const int ar = tid >> 3, acx = (tid & 7) * 4;     // A: rows ar+16i (i<8)
  const int bn = tid & 63, bkq = (tid >> 6) * 4;    // B: n-row bn, k cols bkq+8i
  const float* xA = &x[(size_t)(mBase + ar) * Kd + acx];
  const float* wB = &w[(size_t)bkq * N3 + nBase + bn];

  float4 fa[8];
  float fb[16];
#pragma unroll
  for (int i = 0; i < 8; ++i) fa[i] = *(const float4*)&xA[(size_t)(i * 16) * Kd];
#pragma unroll
  for (int i = 0; i < 4; ++i) {
    const float* s = &wB[(size_t)(i * 8) * N3];
    fb[i*4+0] = s[0]; fb[i*4+1] = s[N3]; fb[i*4+2] = s[2*N3]; fb[i*4+3] = s[3*N3];
  }

  v8f acc[2][4] = {};

  for (int k0 = 0; k0 < Kd; k0 += 32) {
    __syncthreads();
#pragma unroll
    for (int i = 0; i < 8; ++i)
      *(uint2*)&Asm[(ar + i * 16) * 40 + acx] =
          pack4(fa[i].x, fa[i].y, fa[i].z, fa[i].w);
#pragma unroll
    for (int i = 0; i < 4; ++i)
      *(uint2*)&BsmT[bn * 40 + bkq + i * 8] =
          pack4(fb[i*4+0], fb[i*4+1], fb[i*4+2], fb[i*4+3]);
    if (k0 + 32 < Kd) {  // next tile's loads overlap with WMMAs below
      const float* xN = xA + (k0 + 32);
      const float* wN = wB + (size_t)(k0 + 32) * N3;
#pragma unroll
      for (int i = 0; i < 8; ++i) fa[i] = *(const float4*)&xN[(size_t)(i * 16) * Kd];
#pragma unroll
      for (int i = 0; i < 4; ++i) {
        const float* s = &wN[(size_t)(i * 8) * N3];
        fb[i*4+0] = s[0]; fb[i*4+1] = s[N3]; fb[i*4+2] = s[2*N3]; fb[i*4+3] = s[3*N3];
      }
    }
    __syncthreads();

    v16bf a0 = a_frag(&Asm[(wave * 32 + ncol) * 40], half);
    v16bf a1 = a_frag(&Asm[(wave * 32 + 16 + ncol) * 40], half);
#pragma unroll
    for (int c = 0; c < 4; ++c) {
      v16bf b = b_frag(&BsmT[(c * 16 + ncol) * 40], half);
      acc[0][c] = wmma_bf16(a0, b, acc[0][c]);
      acc[1][c] = wmma_bf16(a1, b, acc[1][c]);
    }
  }

  // Epilogue. Block-uniform: tile stays inside one of Q/K/V, one head, one batch.
  const int which = nBase >> 10;              // 0=Q 1=K 2=V
  const int hh = (nBase & 1023) >> 6;
  const int bb = mBase >> 11;
  const int ttBase = (mBase & 2047) + wave * 32 + 8 * half;
  float bv[4];
#pragma unroll
  for (int c = 0; c < 4; ++c) bv[c] = bias[nBase + c * 16 + ncol];

  if (which == 2) {  // V transposed [B,H,D,T]
    __bf16* dst = Vo + ((size_t)bb * H + hh) * D * T + ttBase;
#pragma unroll
    for (int rf = 0; rf < 2; ++rf)
#pragma unroll
      for (int c = 0; c < 4; ++c)
#pragma unroll
        for (int r = 0; r < 8; ++r)
          dst[(size_t)(c * 16 + ncol) * T + rf * 16 + r] =
              (__bf16)(acc[rf][c][r] + bv[c]);
  } else {           // Q/K: [B,H,T,D]
    __bf16* dst = (which == 0 ? Qo : Ko) +
                  (((size_t)bb * H + hh) * T + ttBase) * D + ncol;
    const float scale = (which == 0) ? QSCALE : 1.0f;
#pragma unroll
    for (int rf = 0; rf < 2; ++rf)
#pragma unroll
      for (int c = 0; c < 4; ++c)
#pragma unroll
        for (int r = 0; r < 8; ++r)
          dst[(size_t)(rf * 16 + r) * D + c * 16] =
              (__bf16)((acc[rf][c][r] + bv[c]) * scale);
  }
}

// ===========================================================================
// Kernel 2: flash attention. Grid (T/128, B*H), block 128 (4 waves x 32 rows).
// Double-buffered K/V tiles, async global->LDS staging when available.
// ===========================================================================
__global__ __launch_bounds__(128)
void attn_fwd_kernel(const __bf16* __restrict__ Q, const __bf16* __restrict__ K,
                     const __bf16* __restrict__ V, __bf16* __restrict__ Y) {
  constexpr int T = 2048, D = 64, C = 1024;
  __shared__ __align__(16) __bf16 Ksm[2][32 * 72];   // [key][d], stride 72
  __shared__ __align__(16) __bf16 VsmT[2][64 * 40];  // [d][key], stride 40
  __shared__ __align__(16) __bf16 Psm[4 * 32 * 32];  // per-wave [m][k], stride 32

  const int tid = threadIdx.x, wave = tid >> 5, lane = tid & 31;
  const int half = lane >> 4, ncol = lane & 15;
  const int bh = blockIdx.y;
  const int b = bh >> 4, h = bh & 15;
  const int qBase = blockIdx.x * 128;
  const int qRow0 = qBase + wave * 32;

  const __bf16* Kp  = K + (size_t)bh * T * D;   // [key][d]
  const __bf16* Vtp = V + (size_t)bh * D * T;   // [d][t]
  const __bf16* Qp  = Q + ((size_t)bh * T + qRow0) * D;

  const int kr_ = tid >> 3, kc_ = (tid & 7) * 8;   // K tile rows kr_+16i
  const int vr_ = tid >> 2, vc_ = (tid & 3) * 8;   // V tile rows vr_+32i

  // Q fragments (pre-scaled by QSCALE): 2 row groups x 2 k-chunks.
  v16bf aq[2][2];
#pragma unroll
  for (int rf = 0; rf < 2; ++rf) {
    const __bf16* qr = Qp + (size_t)(rf * 16 + ncol) * D;
    aq[rf][0] = a_frag(qr, half);
    aq[rf][1] = a_frag(qr + 32, half);
  }

  v8f acc[2][4] = {};
  float mrow[2][8], lrow[2][8];
#pragma unroll
  for (int rf = 0; rf < 2; ++rf)
#pragma unroll
    for (int r = 0; r < 8; ++r) { mrow[rf][r] = -1e30f; lrow[rf][r] = 0.0f; }

  const int nkb = (qBase >> 5) + 4;  // 32-key blocks covering keys <= qBase+127
  __bf16* pw = &Psm[wave * 1024];

  // Prologue: stage key-block 0 into buffer 0.
#pragma unroll
  for (int i = 0; i < 2; ++i) {
    copy16(&Ksm[0][(kr_ + 16 * i) * 72 + kc_],
           Kp + (size_t)(kr_ + 16 * i) * D + kc_);
    copy16(&VsmT[0][(vr_ + 32 * i) * 40 + vc_],
           Vtp + (size_t)(vr_ + 32 * i) * T + vc_);
  }
  wait_async();
  __syncthreads();

  for (int jb = 0; jb < nkb; ++jb) {
    const int cur = jb & 1;
    const int kRow0 = jb * 32;
    if (jb + 1 < nkb) {  // async-stage next block into the other buffer
      const int kn = kRow0 + 32, nxt = cur ^ 1;
#pragma unroll
      for (int i = 0; i < 2; ++i) {
        copy16(&Ksm[nxt][(kr_ + 16 * i) * 72 + kc_],
               Kp + (size_t)(kn + kr_ + 16 * i) * D + kc_);
        copy16(&VsmT[nxt][(vr_ + 32 * i) * 40 + vc_],
               Vtp + (size_t)(vr_ + 32 * i) * T + kn + vc_);
      }
    }

    // S = Q K^T : B-fragments shared across both row groups.
    v8f s[2][2];
#pragma unroll
    for (int nb = 0; nb < 2; ++nb) {
      const __bf16* krow = &Ksm[cur][(nb * 16 + ncol) * 72];
      v16bf bk0 = b_frag(krow, half);
      v16bf bk1 = b_frag(krow + 32, half);
      int kg = kRow0 + nb * 16 + ncol;
#pragma unroll
      for (int rf = 0; rf < 2; ++rf) {
        v8f sc = {};
        sc = wmma_bf16(aq[rf][0], bk0, sc);
        sc = wmma_bf16(aq[rf][1], bk1, sc);
#pragma unroll
        for (int r = 0; r < 8; ++r) {
          int qg = qRow0 + rf * 16 + r + 8 * half;
          if (kg > qg) sc[r] = -1e30f;  // causal mask (log2 domain)
        }
        s[rf][nb] = sc;
      }
    }

    // Online softmax in base 2 (scores already scaled by log2e).
#pragma unroll
    for (int rf = 0; rf < 2; ++rf) {
#pragma unroll
      for (int r = 0; r < 8; ++r) {
        float bm = fmaxf(s[rf][0][r], s[rf][1][r]);
#pragma unroll
        for (int off = 1; off < 16; off <<= 1)
          bm = fmaxf(bm, __shfl_xor(bm, off, 32));
        float nm = fmaxf(mrow[rf][r], bm);
        float corr = __builtin_amdgcn_exp2f(mrow[rf][r] - nm);
        float p0 = __builtin_amdgcn_exp2f(s[rf][0][r] - nm);
        float p1 = __builtin_amdgcn_exp2f(s[rf][1][r] - nm);
        float ps = p0 + p1;
#pragma unroll
        for (int off = 1; off < 16; off <<= 1)
          ps += __shfl_xor(ps, off, 32);
        mrow[rf][r] = nm;
        lrow[rf][r] = lrow[rf][r] * corr + ps;
#pragma unroll
        for (int c = 0; c < 4; ++c) acc[rf][c][r] *= corr;
        __bf16* pp = &pw[(rf * 16 + r + 8 * half) * 32];
        pp[ncol] = (__bf16)p0;
        pp[16 + ncol] = (__bf16)p1;
      }
    }

    // Reload P as A-fragments (same-wave LDS RAW).
    v16bf ap0 = a_frag(&pw[ncol * 32], half);
    v16bf ap1 = a_frag(&pw[(16 + ncol) * 32], half);

    // O += P @ V : V-fragments shared across both row groups.
#pragma unroll
    for (int c = 0; c < 4; ++c) {
      v16bf bvf = b_frag(&VsmT[cur][(c * 16 + ncol) * 40], half);
      acc[0][c] = wmma_bf16(ap0, bvf, acc[0][c]);
      acc[1][c] = wmma_bf16(ap1, bvf, acc[1][c]);
    }

    wait_async();     // next buffer filled (per-wave), then block-wide rendezvous
    __syncthreads();
  }

  // Normalize, write Y [B,T,C] bf16 (head h occupies cols h*D..h*D+63).
#pragma unroll
  for (int rf = 0; rf < 2; ++rf)
#pragma unroll
    for (int r = 0; r < 8; ++r) {
      float inv = 1.0f / lrow[rf][r];
      int qg = qRow0 + rf * 16 + r + 8 * half;
      __bf16* yrow = Y + ((size_t)b * T + qg) * C + h * D;
#pragma unroll
      for (int c = 0; c < 4; ++c)
        yrow[c * 16 + ncol] = (__bf16)(acc[rf][c][r] * inv);
    }
}

// ===========================================================================
// Kernel 3: out = Y @ w_proj + b_proj (fp32 out). Grid (M/128, C/64).
// ===========================================================================
__global__ __launch_bounds__(128)
void proj_gemm_kernel(const __bf16* __restrict__ Yb, const float* __restrict__ w,
                      const float* __restrict__ bias, float* __restrict__ out) {
  constexpr int Kd = 1024, N = 1024;
  __shared__ __align__(16) __bf16 Asm[128 * 40];
  __shared__ __align__(16) __bf16 BsmT[64 * 40];

  const int tid = threadIdx.x, wave = tid >> 5, lane = tid & 31;
  const int half = lane >> 4, ncol = lane & 15;
  const int mBase = blockIdx.x * 128, nBase = blockIdx.y * 64;

  const int par = tid >> 2, pac = (tid & 3) * 8;   // A: rows par+32i (i<4)
  const int bn = tid & 63, bkq = (tid >> 6) * 4;
  const __bf16* yA = &Yb[(size_t)(mBase + par) * Kd + pac];
  const float* wB = &w[(size_t)bkq * N + nBase + bn];

  uint4 ya[4];
  float fb[16];
#pragma unroll
  for (int i = 0; i < 4; ++i) ya[i] = *(const uint4*)&yA[(size_t)(i * 32) * Kd];
#pragma unroll
  for (int i = 0; i < 4; ++i) {
    const float* s = &wB[(size_t)(i * 8) * N];
    fb[i*4+0] = s[0]; fb[i*4+1] = s[N]; fb[i*4+2] = s[2*N]; fb[i*4+3] = s[3*N];
  }

  v8f acc[2][4] = {};

  for (int k0 = 0; k0 < Kd; k0 += 32) {
    __syncthreads();
#pragma unroll
    for (int i = 0; i < 4; ++i)
      *(uint4*)&Asm[(par + 32 * i) * 40 + pac] = ya[i];
#pragma unroll
    for (int i = 0; i < 4; ++i)
      *(uint2*)&BsmT[bn * 40 + bkq + i * 8] =
          pack4(fb[i*4+0], fb[i*4+1], fb[i*4+2], fb[i*4+3]);
    if (k0 + 32 < Kd) {
      const __bf16* yN = yA + (k0 + 32);
      const float* wN = wB + (size_t)(k0 + 32) * N;
#pragma unroll
      for (int i = 0; i < 4; ++i) ya[i] = *(const uint4*)&yN[(size_t)(i * 32) * Kd];
#pragma unroll
      for (int i = 0; i < 4; ++i) {
        const float* s = &wN[(size_t)(i * 8) * N];
        fb[i*4+0] = s[0]; fb[i*4+1] = s[N]; fb[i*4+2] = s[2*N]; fb[i*4+3] = s[3*N];
      }
    }
    __syncthreads();

    v16bf a0 = a_frag(&Asm[(wave * 32 + ncol) * 40], half);
    v16bf a1 = a_frag(&Asm[(wave * 32 + 16 + ncol) * 40], half);
#pragma unroll
    for (int c = 0; c < 4; ++c) {
      v16bf b = b_frag(&BsmT[(c * 16 + ncol) * 40], half);
      acc[0][c] = wmma_bf16(a0, b, acc[0][c]);
      acc[1][c] = wmma_bf16(a1, b, acc[1][c]);
    }
  }

  float bv[4];
#pragma unroll
  for (int c = 0; c < 4; ++c) bv[c] = bias[nBase + c * 16 + ncol];
  float* op = &out[(size_t)(mBase + wave * 32 + 8 * half) * N + nBase + ncol];
#pragma unroll
  for (int rf = 0; rf < 2; ++rf)
#pragma unroll
    for (int c = 0; c < 4; ++c)
#pragma unroll
      for (int r = 0; r < 8; ++r)
        op[(size_t)(rf * 16 + r) * N + c * 16] = acc[rf][c][r] + bv[c];
}

// ===========================================================================
extern "C" void kernel_launch(void* const* d_in, const int* in_sizes, int n_in,
                              void* d_out, int out_size, void* d_ws, size_t ws_size,
                              hipStream_t stream) {
  (void)in_sizes; (void)n_in; (void)out_size; (void)ws_size;
  const float* x      = (const float*)d_in[0];
  const float* w_qkv  = (const float*)d_in[1];
  const float* b_qkv  = (const float*)d_in[2];
  const float* w_proj = (const float*)d_in[3];
  const float* b_proj = (const float*)d_in[4];
  float* out = (float*)d_out;

  // Workspace (64 MB): Q,K [B,H,T,D] bf16; V [B,H,D,T] bf16; Y [B,T,C] bf16
  const size_t qkvElems = (size_t)4 * 16 * 2048 * 64;  // 8,388,608
  char* ws = (char*)d_ws;
  __bf16* Q  = (__bf16*)(ws);
  __bf16* K  = (__bf16*)(ws + qkvElems * 2);
  __bf16* V  = (__bf16*)(ws + qkvElems * 4);
  __bf16* Yb = (__bf16*)(ws + qkvElems * 6);

  qkv_gemm_kernel<<<dim3(64, 48), 128, 0, stream>>>(x, w_qkv, b_qkv, Q, K, V);
  attn_fwd_kernel<<<dim3(16, 64), 128, 0, stream>>>(Q, K, V, Yb);
  proj_gemm_kernel<<<dim3(64, 16), 128, 0, stream>>>(Yb, w_proj, b_proj, out);
}
